// non_local_block_4535485464781
// MI455X (gfx1250) — matrix operand order repaired and checked by reading, not compiled, not run on hardware
//
#include <hip/hip_runtime.h>
#include <hip/hip_bf16.h>

// ---------------- types ----------------
typedef __attribute__((ext_vector_type(16))) __bf16          bf16x16;
typedef __attribute__((ext_vector_type(8)))  float           f32x8;
typedef __attribute__((ext_vector_type(4)))  float           f32x4;
typedef __attribute__((ext_vector_type(16))) unsigned short  u16x16;
typedef __attribute__((ext_vector_type(8)))  unsigned short  u16x8;
typedef __attribute__((ext_vector_type(4)))  unsigned short  u16x4;

#define NB   8
#define NN   3136      // H*W
#define CC   256
#define II   128
#define SP   3144      // LDS score-strip stride (bf16 elems), 16B-aligned rows

// fp32 -> bf16 (round to nearest even-ish)
static __device__ __forceinline__ unsigned short f2bf(float f) {
  unsigned u = __float_as_uint(f);
  u += 0x7fffu + ((u >> 16) & 1u);
  return (unsigned short)(u >> 16);
}
static __device__ __forceinline__ float bf2f(unsigned short u) {
  return __uint_as_float(((unsigned)u) << 16);
}

// ---- WMMA fragment loaders (bf16, 16x32 A / 32x16 B per ISA 7.12.2) ----
static __device__ __forceinline__ bf16x16 load_frag_a(const unsigned short* p, int ld, int lane) {
  const unsigned short* r = p + (size_t)(lane & 15) * ld + ((lane >> 4) << 3);
  u16x16 t;
  *(u16x8*)&t       = *(const u16x8*)(r);        // K = hi*8 .. +7
  *((u16x8*)&t + 1) = *(const u16x8*)(r + 16);   // K = 16+hi*8 .. +7
  return __builtin_bit_cast(bf16x16, t);
}
static __device__ __forceinline__ bf16x16 load_frag_b(const unsigned short* p, int ld, int lane) {
  const unsigned short* r = p + (size_t)((lane >> 4) << 3) * ld + (lane & 15);  // lane = N-col, K = hi*8.. and 16+hi*8..
  u16x16 t;
#pragma unroll
  for (int j = 0; j < 8; ++j) { t[j] = r[(size_t)j * ld]; t[8 + j] = r[(size_t)(16 + j) * ld]; }
  return __builtin_bit_cast(bf16x16, t);
}
static __device__ __forceinline__ f32x8 wmma_bf16(bf16x16 a, bf16x16 b, f32x8 c) {
  return __builtin_amdgcn_wmma_f32_16x16x32_bf16(false, a, false, b, (short)0, c, false, false);
}

// =====================================================================
// Kernel 0: one-shot fp32 -> bf16 conversion (x and the 3 weights).
// =====================================================================
__global__ __launch_bounds__(256) void nlb_cvt_kernel(const float* __restrict__ src,
                                                      unsigned short* __restrict__ dst,
                                                      int n8) {
  const int i = blockIdx.x * 256 + threadIdx.x;
  if (i >= n8) return;
  const float* s = src + (size_t)i * 8;
  u16x8 v;
#pragma unroll
  for (int j = 0; j < 8; ++j) v[j] = f2bf(s[j]);
  *(u16x8*)(dst + (size_t)i * 8) = v;
}

// =====================================================================
// Kernel 1: theta/phi/g projections from pre-converted bf16 x/weights.
// 8 waves/block, 16 rows/wave. Epilogue staged via wave-private LDS slab
// so global stores are coalesced b128 instead of scattered b16.
// =====================================================================
template<int MODE>  // 0: row-major [B*N][I]; 1: transposed phiT [B][I][N]
static __device__ __forceinline__ void proj_one(
    const unsigned short* __restrict__ Wb, const bf16x16* ax,
    unsigned short* __restrict__ dst, unsigned short* st /*wave-private LDS, 2048 elems*/,
    size_t row0, int b, int nb, int lane)
{
  const int hi = lane >> 4, nl = lane & 15;
#pragma unroll
  for (int it = 0; it < 8; ++it) {          // I = 128 -> 8 column tiles
    bf16x16 bw[8];
#pragma unroll
    for (int kk = 0; kk < 8; ++kk)          // batch all K loads -> one clause
      bw[kk] = load_frag_b(Wb + (size_t)(kk * 32) * II + it * 16, II, lane);
    f32x8 acc = {0.f, 0.f, 0.f, 0.f, 0.f, 0.f, 0.f, 0.f};
#pragma unroll
    for (int kk = 0; kk < 8; ++kk)
      acc = wmma_bf16(ax[kk], bw[kk], acc);
#pragma unroll
    for (int r = 0; r < 8; ++r) {           // stage tile into LDS slab
      const int m = r + 8 * hi;
      const int i = it * 16 + nl;
      const unsigned short v = f2bf(acc[r]);
      if (MODE == 0) st[m * II + i] = v;    // [16 rows][128 cols]
      else           st[i * 16 + m] = v;    // [128 i][16 n]
    }
  }
  // flush slab to global, coalesced (same-wave DS ordering: no barrier needed)
  if (MODE == 0) {
    unsigned short* d = dst + row0 * II;    // 16*128 contiguous elements
#pragma unroll
    for (int q = 0; q < 8; ++q)
      *(u16x8*)(d + lane * 64 + q * 8) = *(const u16x8*)(st + lane * 64 + q * 8);
  } else {
#pragma unroll
    for (int r4 = 0; r4 < 4; ++r4) {
      const int i = lane * 4 + r4;
      unsigned short* d = dst + ((size_t)b * II + i) * NN + nb;
      *(u16x8*)(d)     = *(const u16x8*)(st + i * 16);
      *(u16x8*)(d + 8) = *(const u16x8*)(st + i * 16 + 8);
    }
  }
}

__global__ __launch_bounds__(256) void nlb_proj_kernel(
    const unsigned short* __restrict__ xb, const unsigned short* __restrict__ wtb,
    const unsigned short* __restrict__ wpb, const unsigned short* __restrict__ wgb,
    unsigned short* __restrict__ theta, unsigned short* __restrict__ phiT,
    unsigned short* __restrict__ g)
{
  __shared__ unsigned short stage[8][16 * II];   // 4 KB per wave, 32 KB total
  const int lane = threadIdx.x & 31;
  const int wave = threadIdx.x >> 5;
  const size_t row0 = ((size_t)blockIdx.x * 8 + wave) * 16;   // flat row in [0, B*N)
  const int   b  = (int)(row0 / NN);
  const int   nb = (int)(row0 - (size_t)b * NN);

  bf16x16 ax[8];
#pragma unroll
  for (int kk = 0; kk < 8; ++kk)
    ax[kk] = load_frag_a(xb + row0 * CC + kk * 32, CC, lane);

  proj_one<0>(wtb, ax, theta, stage[wave], row0, b, nb, lane);
  proj_one<1>(wpb, ax, phiT,  stage[wave], row0, b, nb, lane);
  proj_one<0>(wgb, ax, g,     stage[wave], row0, b, nb, lane);
}

// =====================================================================
// Kernel 2: per (16-row strip, batch): S = theta*phiT (WMMA, bf16 strip
// in LDS) -> softmax (fp32) -> write w (float4) -> P bf16 in place ->
// O = P*g (WMMA) -> out = O*w_o + x (float4).
// =====================================================================
__global__ __launch_bounds__(256) void nlb_attn_kernel(
    const unsigned short* __restrict__ theta, const unsigned short* __restrict__ phiT,
    const unsigned short* __restrict__ g, const float* __restrict__ w_o,
    const float* __restrict__ x, float* __restrict__ out, float* __restrict__ wfull)
{
  __shared__ unsigned short S[16 * SP];     // bf16 scores, then P, ~101 KB
  __shared__ float Ored[16 * II];           // 8 KB fp32 O accumulator
  __shared__ float red[256];
  __shared__ float rowmax[16];
  __shared__ float rinv[16];

  const int tid  = threadIdx.x;
  const int lane = tid & 31;
  const int wave = tid >> 5;
  const int b    = blockIdx.y;
  const int n0   = blockIdx.x * 16;
  const size_t gr = (size_t)b * NN + n0;

  for (int idx = tid; idx < 16 * II; idx += 256) Ored[idx] = 0.f;

  bf16x16 at[4];
#pragma unroll
  for (int kk = 0; kk < 4; ++kk)
    at[kk] = load_frag_a(theta + gr * II + kk * 32, II, lane);

  // -------- Pass A: scores S[16][3136] (bf16), double-buffered phi frags ----
  const unsigned short* phiB = phiT + (size_t)b * II * NN;
  const int hi = lane >> 4, nl = lane & 15;
  {
    const int JT = NN / 16;                  // 196 column tiles
    int j = wave;
    bf16x16 cur[4];
#pragma unroll
    for (int kk = 0; kk < 4; ++kk)
      cur[kk] = load_frag_b(phiB + (size_t)(kk * 32) * NN + j * 16, NN, lane);
    while (j < JT) {
      const int jn = j + 8;
      bf16x16 nxt[4] = {cur[0], cur[1], cur[2], cur[3]};
      if (jn < JT) {
#pragma unroll
        for (int kk = 0; kk < 4; ++kk)       // issue next tile's loads early
          nxt[kk] = load_frag_b(phiB + (size_t)(kk * 32) * NN + jn * 16, NN, lane);
      }
      f32x8 acc = {0.f, 0.f, 0.f, 0.f, 0.f, 0.f, 0.f, 0.f};
#pragma unroll
      for (int kk = 0; kk < 4; ++kk)
        acc = wmma_bf16(at[kk], cur[kk], acc);
#pragma unroll
      for (int r = 0; r < 8; ++r)
        S[(r + 8 * hi) * SP + j * 16 + nl] = f2bf(acc[r]);
#pragma unroll
      for (int kk = 0; kk < 4; ++kk) cur[kk] = nxt[kk];
      j = jn;
    }
  }
  __syncthreads();

  // -------- Pass B: row max then exp-sum (16 rows x 16 threads, 4-wide) ----
  {
    const int row = tid >> 4, j0 = tid & 15;
    float mx = -3.0e38f;
    for (int c4 = j0; c4 < NN / 4; c4 += 16) {
      u16x4 v = *(const u16x4*)(S + row * SP + c4 * 4);
      mx = fmaxf(mx, fmaxf(fmaxf(bf2f(v[0]), bf2f(v[1])),
                           fmaxf(bf2f(v[2]), bf2f(v[3]))));
    }
    red[tid] = mx;
    __syncthreads();
    if (tid < 16) {
      float m = red[tid * 16];
#pragma unroll
      for (int k = 1; k < 16; ++k) m = fmaxf(m, red[tid * 16 + k]);
      rowmax[tid] = m;
    }
    __syncthreads();
    const float m = rowmax[row];
    float sm = 0.f;
    for (int c4 = j0; c4 < NN / 4; c4 += 16) {
      u16x4 v = *(const u16x4*)(S + row * SP + c4 * 4);
      sm += __expf(bf2f(v[0]) - m) + __expf(bf2f(v[1]) - m) +
            __expf(bf2f(v[2]) - m) + __expf(bf2f(v[3]) - m);
    }
    red[tid] = sm;
    __syncthreads();
    if (tid < 16) {
      float s = 0.f;
#pragma unroll
      for (int k = 0; k < 16; ++k) s += red[tid * 16 + k];
      rinv[tid] = 1.0f / s;
    }
    __syncthreads();
  }

  // -------- Pass C: write normalized w (float4); P (bf16) in place --------
  for (int r = 0; r < 16; ++r) {
    const float m = rowmax[r], rs = rinv[r];
    float* wrow = wfull + (gr + r) * (size_t)NN;
    for (int c4 = tid; c4 < NN / 4; c4 += 256) {
      u16x4 v = *(const u16x4*)(S + r * SP + c4 * 4);
      f32x4 e;
#pragma unroll
      for (int j = 0; j < 4; ++j) e[j] = __expf(bf2f(v[j]) - m) * rs;
      *(f32x4*)(wrow + c4 * 4) = e;            // coalesced 16B stores of w
      u16x4 pv;
#pragma unroll
      for (int j = 0; j < 4; ++j) pv[j] = f2bf(e[j]);
      *(u16x4*)(S + r * SP + c4 * 4) = pv;     // in-place P, same slot per thread
    }
  }
  __syncthreads();

  // -------- Pass D: O += P * g  (K over 3136 columns, 98 k-steps) --------
  f32x8 accO[8];
#pragma unroll
  for (int it = 0; it < 8; ++it)
    accO[it] = (f32x8){0.f, 0.f, 0.f, 0.f, 0.f, 0.f, 0.f, 0.f};

  const unsigned short* gB = g + (size_t)b * NN * II;
  for (int kt = wave; kt < NN / 32; kt += 8) {
    bf16x16 ap = load_frag_a(S + kt * 32, SP, lane);   // ds_load from LDS strip
    bf16x16 bg[8];
#pragma unroll
    for (int it = 0; it < 8; ++it)                     // batch 8 frag loads -> clause
      bg[it] = load_frag_b(gB + (size_t)(kt * 32) * II + it * 16, II, lane);
#pragma unroll
    for (int it = 0; it < 8; ++it)
      accO[it] = wmma_bf16(ap, bg[it], accO[it]);
  }
  // cross-wave reduction into LDS via ds_add_f32
#pragma unroll
  for (int it = 0; it < 8; ++it)
#pragma unroll
    for (int r = 0; r < 8; ++r)
      atomicAdd(&Ored[(r + 8 * hi) * II + it * 16 + nl], accO[it][r]);
  __syncthreads();

  // -------- Pass E: out = O * w_o + x  (16x256, fp32 VALU, float4) --------
  for (int idx4 = tid; idx4 < 16 * (CC / 4); idx4 += 256) {
    const int row = idx4 >> 6;
    const int c   = (idx4 & 63) * 4;
    f32x4 a = {0.f, 0.f, 0.f, 0.f};
#pragma unroll 4
    for (int i = 0; i < II; ++i) {
      const float o = Ored[row * II + i];
      f32x4 wv = *(const f32x4*)(w_o + i * CC + c);
#pragma unroll
      for (int j = 0; j < 4; ++j) a[j] = fmaf(o, wv[j], a[j]);
    }
    f32x4 xv = *(const f32x4*)(x + (gr + row) * CC + c);
#pragma unroll
    for (int j = 0; j < 4; ++j) a[j] += xv[j];
    *(f32x4*)(out + (gr + row) * CC + c) = a;
  }
}

// =====================================================================
extern "C" void kernel_launch(void* const* d_in, const int* in_sizes, int n_in,
                              void* d_out, int out_size, void* d_ws, size_t ws_size,
                              hipStream_t stream) {
  const float* x       = (const float*)d_in[0];
  const float* w_theta = (const float*)d_in[1];
  const float* w_phi   = (const float*)d_in[2];
  const float* w_g     = (const float*)d_in[3];
  const float* w_o     = (const float*)d_in[4];

  const size_t PROJ = (size_t)NB * NN * II;   // 3,211,264
  const size_t XE   = (size_t)NB * NN * CC;   // 6,422,528
  const size_t WE   = (size_t)CC * II;        // 32,768

  unsigned short* theta = (unsigned short*)d_ws;
  unsigned short* phiT  = theta + PROJ;
  unsigned short* g     = phiT + PROJ;
  unsigned short* xb    = g + PROJ;
  unsigned short* wtb   = xb + XE;
  unsigned short* wpb   = wtb + WE;
  unsigned short* wgb   = wpb + WE;

  float* out   = (float*)d_out;
  float* wfull = out + XE;                    // tuple: (out, w) concatenated

  nlb_cvt_kernel<<<(int)(XE / 8 + 255) / 256, 256, 0, stream>>>(x, xb, (int)(XE / 8));
  nlb_cvt_kernel<<<(int)(WE / 8 + 255) / 256, 256, 0, stream>>>(w_theta, wtb, (int)(WE / 8));
  nlb_cvt_kernel<<<(int)(WE / 8 + 255) / 256, 256, 0, stream>>>(w_phi,  wpb, (int)(WE / 8));
  nlb_cvt_kernel<<<(int)(WE / 8 + 255) / 256, 256, 0, stream>>>(w_g,    wgb, (int)(WE / 8));

  nlb_proj_kernel<<<(NB * NN) / 128, 256, 0, stream>>>(xb, wtb, wpb, wgb, theta, phiT, g);
  nlb_attn_kernel<<<dim3(NN / 16, NB), 256, 0, stream>>>(theta, phiT, g, w_o, x, out, wfull);
}
